// ProjectClass_38259568673435
// MI455X (gfx1250) — matrix operand order repaired and checked by reading, not compiled, not run on hardware
//
#include <hip/hip_runtime.h>
#include <hip/hip_bf16.h>

// ---------------------------------------------------------------------------
// Problem dimensions (fixed by the reference)
// ---------------------------------------------------------------------------
#define U_DIM   512
#define D4      2048          // 4*U
#define D2      1024          // 2*U
#define C_IN    768
#define BSZ     4096
#define NOY     2048
#define BN_EPS  1e-5f
#define COS_EPS 1e-6f

// ---------------------------------------------------------------------------
// WMMA types (CDNA5 wave32)
// ---------------------------------------------------------------------------
typedef __attribute__((ext_vector_type(16))) __bf16 v16bf;
typedef __attribute__((ext_vector_type(8)))  float  v8f;

union FragBF {
    unsigned int u[8];
    v16bf        v;
};

__device__ __forceinline__ unsigned short f32_to_bf16(float f) {
    unsigned int u = __float_as_uint(f);
    // round-to-nearest-even
    unsigned int r = (u + 0x7fffu + ((u >> 16) & 1u)) >> 16;
    return (unsigned short)r;
}

// CDNA5 async global->LDS copy (ASYNCcnt-tracked, no VGPR round trip).
// lds_addr: LDS byte address (low 32 bits of the generic __shared__ pointer).
__device__ __forceinline__ void async_copy_b128(unsigned lds_addr, const void* gptr) {
    asm volatile("global_load_async_to_lds_b128 %0, %1, off"
                 :
                 : "v"(lds_addr), "v"((unsigned long long)(uintptr_t)gptr)
                 : "memory");
}

__device__ __forceinline__ void wait_async0() {
    asm volatile("s_wait_asynccnt 0" ::: "memory");
}

// ---------------------------------------------------------------------------
// BatchNorm pass 1: per-column partial sums over a chunk of rows.
// grid = (cols/128, nChunks), block = 128.
// ---------------------------------------------------------------------------
__global__ __launch_bounds__(128) void bn_partial_kernel(
    const float* __restrict__ X, int cols, int rowsPerChunk,
    float* __restrict__ psum, float* __restrict__ psumsq)
{
    int c  = blockIdx.x * 128 + threadIdx.x;
    int r0 = blockIdx.y * rowsPerChunk;
    float s = 0.f, q = 0.f;
    for (int r = 0; r < rowsPerChunk; ++r) {
        float v = X[(size_t)(r0 + r) * cols + c];
        s += v;
        q += v * v;
    }
    psum[(size_t)blockIdx.y * cols + c]   = s;
    psumsq[(size_t)blockIdx.y * cols + c] = q;
}

// BatchNorm pass 2: combine chunks, fold gamma/beta into scale/shift.
__global__ __launch_bounds__(128) void bn_finalize_kernel(
    const float* __restrict__ psum, const float* __restrict__ psumsq,
    const float* __restrict__ g, const float* __restrict__ b,
    float* __restrict__ scale, float* __restrict__ shift,
    int cols, int nChunks, float invRows)
{
    int c = blockIdx.x * 128 + threadIdx.x;
    float s = 0.f, q = 0.f;
    for (int k = 0; k < nChunks; ++k) {
        s += psum[(size_t)k * cols + c];
        q += psumsq[(size_t)k * cols + c];
    }
    float mean = s * invRows;
    float var  = q * invRows - mean * mean;   // biased variance
    float inv  = rsqrtf(var + BN_EPS);
    float sc   = g[c] * inv;
    scale[c] = sc;
    shift[c] = b[c] - mean * sc;
}

// BatchNorm apply + convert to bf16.
__global__ __launch_bounds__(256) void bn_apply_kernel(
    const float* __restrict__ X,
    const float* __restrict__ scale, const float* __restrict__ shift,
    unsigned short* __restrict__ out, int cols, size_t total)
{
    size_t idx = (size_t)blockIdx.x * 256 + threadIdx.x;
    if (idx >= total) return;
    int c = (int)(idx % cols);
    out[idx] = f32_to_bf16(X[idx] * scale[c] + shift[c]);
}

// Weight convert f32 [K][N] -> bf16 transposed [N][K]  (NT GEMM layout).
__global__ __launch_bounds__(256) void wt_convert_kernel(
    const float* __restrict__ W, unsigned short* __restrict__ Wt,
    int K, int N)
{
    size_t idx = (size_t)blockIdx.x * 256 + threadIdx.x;
    if (idx >= (size_t)K * N) return;
    int n = (int)(idx % N);
    int k = (int)(idx / N);
    Wt[(size_t)n * K + k] = f32_to_bf16(W[idx]);   // coalesced read, scattered write
}

// ---------------------------------------------------------------------------
// Tiled bf16 WMMA NT-GEMM:  C[M][N] = A[M][K] * Bt[N][K]^T  (f32 accumulate)
//   block = 256 threads (8 wave32), tile = 128x128, K-step = 32
//   double-buffered LDS staged via global_load_async_to_lds_b128 (ASYNCcnt)
//   each wave: 2 (M) x 4 (N) fragments of 16x16, v_wmma_f32_16x16x32_bf16
// Epilogues:
//   EPI==0 : bf16( relu(acc + bias[col]) )            -> Cbf
//   EPI==1 : acc + bias[col]                          -> Cf and bf16 -> Cbf
//   EPI==2 : acc / max(rowScale[row]*colScale[col],e) -> Cf   (cosine score)
// ---------------------------------------------------------------------------
template <int EPI>
__global__ __launch_bounds__(256) void gemm_bf16_kernel(
    const unsigned short* __restrict__ A,    // [M][K] bf16
    const unsigned short* __restrict__ Bt,   // [N][K] bf16
    int M, int N, int K,
    const float* __restrict__ bias,          // [N] (EPI 0/1)
    float* __restrict__ Cf,                  // [M][N] (EPI 1/2)
    unsigned short* __restrict__ Cbf,        // [M][N] (EPI 0/1)
    const float* __restrict__ rowScale,      // [M] (EPI 2)
    const float* __restrict__ colScale)      // [N] (EPI 2)
{
    constexpr int BK  = 32;
    constexpr int LDK = BK + 8;              // pad to dodge bank conflicts

    __shared__ unsigned short As[2][128][LDK];
    __shared__ unsigned short Bs[2][128][LDK];
    constexpr unsigned BUF_STRIDE = 128 * LDK * 2;   // bytes per buffer

    const int    tid     = threadIdx.x;
    const int    lane    = tid & 31;
    const int    wave    = tid >> 5;
    const int    l15     = lane & 15;
    const int    half    = lane >> 4;
    const int    wm      = (wave >> 1) * 32;  // wave row offset in tile
    const int    wn      = (wave & 1) * 64;   // wave col offset in tile
    const size_t rowBase = (size_t)blockIdx.y * 128;
    const size_t colBase = (size_t)blockIdx.x * 128;

    // ---- per-thread staging slots: 2 x 16B chunks per matrix per tile ----
    const int rC0 = tid >> 2;                    // chunk0 row    (0..63)
    const int rC1 = (tid + 256) >> 2;            // chunk1 row    (64..127)
    const int cO0 = (tid & 3) * 8;               // chunk0 short offset
    const int cO1 = cO0;                         // (tid+256)&3 == tid&3
    const unsigned short* gA0 = A  + (rowBase + rC0) * (size_t)K + cO0;
    const unsigned short* gA1 = A  + (rowBase + rC1) * (size_t)K + cO1;
    const unsigned short* gB0 = Bt + (colBase + rC0) * (size_t)K + cO0;
    const unsigned short* gB1 = Bt + (colBase + rC1) * (size_t)K + cO1;
    const unsigned ldsA0 = (unsigned)(uintptr_t)&As[0][rC0][cO0];
    const unsigned ldsA1 = (unsigned)(uintptr_t)&As[0][rC1][cO1];
    const unsigned ldsB0 = (unsigned)(uintptr_t)&Bs[0][rC0][cO0];
    const unsigned ldsB1 = (unsigned)(uintptr_t)&Bs[0][rC1][cO1];

    auto stage = [&](int kt, int buf) {
        const int      k0   = kt * BK;
        const unsigned boff = buf ? BUF_STRIDE : 0u;
        async_copy_b128(ldsA0 + boff, gA0 + k0);
        async_copy_b128(ldsA1 + boff, gA1 + k0);
        async_copy_b128(ldsB0 + boff, gB0 + k0);
        async_copy_b128(ldsB1 + boff, gB1 + k0);
    };

    v8f acc[2][4];
#pragma unroll
    for (int fm = 0; fm < 2; ++fm)
#pragma unroll
        for (int fn = 0; fn < 4; ++fn)
            acc[fm][fn] = (v8f){0.f, 0.f, 0.f, 0.f, 0.f, 0.f, 0.f, 0.f};

    const int ktiles = K / BK;

    stage(0, 0);
    wait_async0();
    __syncthreads();

    for (int kt = 0; kt < ktiles; ++kt) {
        const int buf = kt & 1;

        // Kick off the next tile into the alternate buffer (no wait yet).
        if (kt + 1 < ktiles) stage(kt + 1, buf ^ 1);

        // A fragment (16x32 bf16, wave32 layout):
        //   lanes 0-15: M=lane; VGPR v<4 -> K=2v..2v+1, v>=4 -> K=16+2(v-4)..
        //   lanes 16-31 shift K by +8 within each half.
        FragBF a[2];
#pragma unroll
        for (int fm = 0; fm < 2; ++fm)
#pragma unroll
            for (int v = 0; v < 8; ++v) {
                int k = ((v < 4) ? 0 : 16) + half * 8 + (v & 3) * 2;
                a[fm].u[v] = *(const unsigned int*)&As[buf][wm + fm * 16 + l15][k];
            }

        // B fragment (32x16 bf16): lane holds N=lane&15; K layout same as A (v<4 -> half*8+2v, v>=4 -> 16+half*8+2(v-4)).
        FragBF b[4];
#pragma unroll
        for (int fn = 0; fn < 4; ++fn)
#pragma unroll
            for (int v = 0; v < 8; ++v) {
                int k = ((v < 4) ? 0 : 16) + half * 8 + (v & 3) * 2;
                b[fn].u[v] = *(const unsigned int*)&Bs[buf][wn + fn * 16 + l15][k];
            }

#pragma unroll
        for (int fm = 0; fm < 2; ++fm)
#pragma unroll
            for (int fn = 0; fn < 4; ++fn)
                acc[fm][fn] = __builtin_amdgcn_wmma_f32_16x16x32_bf16(
                    false, a[fm].v, false, b[fn].v,
                    (short)0, acc[fm][fn], false, false);

        // Make the next buffer visible to everyone before using it.
        if (kt + 1 < ktiles) {
            wait_async0();
            __syncthreads();
        }
    }

    // Epilogue. C/D layout: lane L holds N=L&15; VGPR r -> M = r + (L>>4)*8.
#pragma unroll
    for (int fm = 0; fm < 2; ++fm) {
#pragma unroll
        for (int fn = 0; fn < 4; ++fn) {
            size_t col = colBase + wn + fn * 16 + l15;
            float  bv  = 0.f, cs = 0.f;
            if (EPI == 0 || EPI == 1) bv = bias[col];
            if (EPI == 2)             cs = colScale[col];
#pragma unroll
            for (int r = 0; r < 8; ++r) {
                size_t row = rowBase + wm + fm * 16 + half * 8 + r;
                float  v   = acc[fm][fn][r];
                if (EPI == 0) {
                    v += bv;
                    v = fmaxf(v, 0.f);
                    Cbf[row * N + col] = f32_to_bf16(v);
                } else if (EPI == 1) {
                    v += bv;
                    Cf[row * N + col]  = v;
                    Cbf[row * N + col] = f32_to_bf16(v);
                } else {
                    float d = fmaxf(rowScale[row] * cs, COS_EPS);
                    Cf[row * N + col] = v / d;
                }
            }
        }
    }
}

// ---------------------------------------------------------------------------
// Row L2-norm: one block per row.
// ---------------------------------------------------------------------------
__global__ __launch_bounds__(256) void row_norm_kernel(
    const float* __restrict__ X, float* __restrict__ out, int cols)
{
    __shared__ float red[256];
    int          row = blockIdx.x;
    const float* xr  = X + (size_t)row * cols;
    float        s   = 0.f;
    for (int c = threadIdx.x; c < cols; c += 256) {
        float v = xr[c];
        s += v * v;
    }
    red[threadIdx.x] = s;
    __syncthreads();
    for (int off = 128; off > 0; off >>= 1) {
        if (threadIdx.x < off) red[threadIdx.x] += red[threadIdx.x + off];
        __syncthreads();
    }
    if (threadIdx.x == 0) out[row] = sqrtf(red[0]);
}

// ---------------------------------------------------------------------------
// Row argmax (first max wins, matching jnp.argmax): one block per row.
// ---------------------------------------------------------------------------
__global__ __launch_bounds__(256) void argmax_kernel(
    const float* __restrict__ score, float* __restrict__ pred, int cols)
{
    __shared__ float bv[256];
    __shared__ int   bi[256];
    int          row = blockIdx.x;
    const float* s   = score + (size_t)row * cols;
    float best = -__builtin_inff();
    int   bidx = 0;
    for (int c = threadIdx.x; c < cols; c += 256) {
        float v = s[c];
        if (v > best) { best = v; bidx = c; }
    }
    bv[threadIdx.x] = best;
    bi[threadIdx.x] = bidx;
    __syncthreads();
    for (int off = 128; off > 0; off >>= 1) {
        if (threadIdx.x < off) {
            float ov = bv[threadIdx.x + off];
            int   oi = bi[threadIdx.x + off];
            if (ov > bv[threadIdx.x] ||
                (ov == bv[threadIdx.x] && oi < bi[threadIdx.x])) {
                bv[threadIdx.x] = ov;
                bi[threadIdx.x] = oi;
            }
        }
        __syncthreads();
    }
    if (threadIdx.x == 0) pred[row] = (float)bi[0];
}

// ---------------------------------------------------------------------------
// Host-side pipeline
// ---------------------------------------------------------------------------
extern "C" void kernel_launch(void* const* d_in, const int* in_sizes, int n_in,
                              void* d_out, int out_size, void* d_ws, size_t ws_size,
                              hipStream_t stream) {
    (void)in_sizes; (void)n_in; (void)out_size; (void)ws_size;

    const float* Hx  = (const float*)d_in[0];   // [BSZ, D4]
    const float* Hy  = (const float*)d_in[1];   // [NOY, C_IN]
    const float* gU  = (const float*)d_in[2];
    const float* bU  = (const float*)d_in[3];
    const float* W1u = (const float*)d_in[4];   // [D4, D2]
    const float* b1u = (const float*)d_in[5];
    const float* W2u = (const float*)d_in[6];   // [D2, D2]
    const float* b2u = (const float*)d_in[7];
    const float* gC  = (const float*)d_in[8];
    const float* bC  = (const float*)d_in[9];
    const float* W1c = (const float*)d_in[10];  // [C_IN, D2]
    const float* b1c = (const float*)d_in[11];
    const float* W2c = (const float*)d_in[12];  // [D2, D2]
    const float* b2c = (const float*)d_in[13];
    // d_in[14] = truthY (unused by the outputs)

    float* score = (float*)d_out;                          // [BSZ, NOY]
    float* pred  = (float*)d_out + (size_t)BSZ * NOY;      // [BSZ]

    // ---- workspace carve-out (256B aligned) ----
    char*  ws  = (char*)d_ws;
    size_t off = 0;
    auto alloc = [&](size_t bytes) -> void* {
        void* p = ws + off;
        off += (bytes + 255) & ~(size_t)255;
        return p;
    };
    unsigned short* Hxn   = (unsigned short*)alloc((size_t)BSZ * D4 * 2);
    unsigned short* Hyn   = (unsigned short*)alloc((size_t)NOY * C_IN * 2);
    unsigned short* W1uT  = (unsigned short*)alloc((size_t)D2 * D4 * 2);
    unsigned short* W2uT  = (unsigned short*)alloc((size_t)D2 * D2 * 2);
    unsigned short* W1cT  = (unsigned short*)alloc((size_t)D2 * C_IN * 2);
    unsigned short* W2cT  = (unsigned short*)alloc((size_t)D2 * D2 * 2);
    unsigned short* H1u   = (unsigned short*)alloc((size_t)BSZ * D2 * 2);
    unsigned short* H1c   = (unsigned short*)alloc((size_t)NOY * D2 * 2);
    float*          txf   = (float*)alloc((size_t)BSZ * D2 * 4);
    unsigned short* txb   = (unsigned short*)alloc((size_t)BSZ * D2 * 2);
    float*          tyf   = (float*)alloc((size_t)NOY * D2 * 4);
    unsigned short* tyb   = (unsigned short*)alloc((size_t)NOY * D2 * 2);
    float*          nx    = (float*)alloc((size_t)BSZ * 4);
    float*          ny    = (float*)alloc((size_t)NOY * 4);
    float*          psum  = (float*)alloc((size_t)32 * D4 * 4);  // reused for Hy
    float*          psq   = (float*)alloc((size_t)32 * D4 * 4);
    float*          scU   = (float*)alloc((size_t)D4 * 4);
    float*          shU   = (float*)alloc((size_t)D4 * 4);
    float*          scC   = (float*)alloc((size_t)C_IN * 4);
    float*          shC   = (float*)alloc((size_t)C_IN * 4);

    // ---- BatchNorm(Hx) -> bf16 ----
    bn_partial_kernel<<<dim3(D4 / 128, 32), 128, 0, stream>>>(Hx, D4, BSZ / 32, psum, psq);
    bn_finalize_kernel<<<D4 / 128, 128, 0, stream>>>(psum, psq, gU, bU, scU, shU, D4, 32, 1.0f / BSZ);
    bn_apply_kernel<<<(unsigned)(((size_t)BSZ * D4 + 255) / 256), 256, 0, stream>>>(
        Hx, scU, shU, Hxn, D4, (size_t)BSZ * D4);

    // ---- BatchNorm(Hy) -> bf16 ----
    bn_partial_kernel<<<dim3(C_IN / 128, 16), 128, 0, stream>>>(Hy, C_IN, NOY / 16, psum, psq);
    bn_finalize_kernel<<<C_IN / 128, 128, 0, stream>>>(psum, psq, gC, bC, scC, shC, C_IN, 16, 1.0f / NOY);
    bn_apply_kernel<<<(unsigned)(((size_t)NOY * C_IN + 255) / 256), 256, 0, stream>>>(
        Hy, scC, shC, Hyn, C_IN, (size_t)NOY * C_IN);

    // ---- weights -> bf16 transposed [N][K] ----
    wt_convert_kernel<<<(unsigned)(((size_t)D4 * D2 + 255) / 256), 256, 0, stream>>>(W1u, W1uT, D4, D2);
    wt_convert_kernel<<<(unsigned)(((size_t)D2 * D2 + 255) / 256), 256, 0, stream>>>(W2u, W2uT, D2, D2);
    wt_convert_kernel<<<(unsigned)(((size_t)C_IN * D2 + 255) / 256), 256, 0, stream>>>(W1c, W1cT, C_IN, D2);
    wt_convert_kernel<<<(unsigned)(((size_t)D2 * D2 + 255) / 256), 256, 0, stream>>>(W2c, W2cT, D2, D2);

    // ---- MLP tower x: H1u = relu(Hxn@W1u + b1u); tx = H1u@W2u + b2u ----
    gemm_bf16_kernel<0><<<dim3(D2 / 128, BSZ / 128), 256, 0, stream>>>(
        Hxn, W1uT, BSZ, D2, D4, b1u, nullptr, H1u, nullptr, nullptr);
    gemm_bf16_kernel<1><<<dim3(D2 / 128, BSZ / 128), 256, 0, stream>>>(
        H1u, W2uT, BSZ, D2, D2, b2u, txf, txb, nullptr, nullptr);

    // ---- MLP tower y ----
    gemm_bf16_kernel<0><<<dim3(D2 / 128, NOY / 128), 256, 0, stream>>>(
        Hyn, W1cT, NOY, D2, C_IN, b1c, nullptr, H1c, nullptr, nullptr);
    gemm_bf16_kernel<1><<<dim3(D2 / 128, NOY / 128), 256, 0, stream>>>(
        H1c, W2cT, NOY, D2, D2, b2c, tyf, tyb, nullptr, nullptr);

    // ---- row norms ----
    row_norm_kernel<<<BSZ, 256, 0, stream>>>(txf, nx, D2);
    row_norm_kernel<<<NOY, 256, 0, stream>>>(tyf, ny, D2);

    // ---- score = (tx @ ty^T) / max(nx*ny, eps) ----
    gemm_bf16_kernel<2><<<dim3(NOY / 128, BSZ / 128), 256, 0, stream>>>(
        txb, tyb, BSZ, NOY, D2, nullptr, score, nullptr, nx, ny);

    // ---- pred = argmax(score, axis=1) ----
    argmax_kernel<<<BSZ, 256, 0, stream>>>(score, pred, NOY);
}
